// GraphNeuralNet_37185826849006
// MI455X (gfx1250) — matrix operand order repaired and checked by reading, not compiled, not run on hardware
//
#include <hip/hip_runtime.h>

#define NN 50000
#define NE 800000
#define IND 128
#define EH 32
#define MH 64

typedef __bf16 bf16;
typedef __attribute__((ext_vector_type(16))) __bf16 v16bf;
typedef __attribute__((ext_vector_type(8)))  float  v8f;

// ---- WMMA fragment swizzle helpers (CDNA5 ISA §7.12.2) ----
// A matrix 16x32 bf16: lane = (half<<4)|m ; element e in lane maps to
//   K = (e<8 ? 8*half + e : 16 + 8*half + (e-8)),  half = (kk>>3)&1
__device__ __forceinline__ int a_off(int kk, int m) {
  int half = (kk >> 3) & 1;
  int e = (kk & 7) | ((kk >> 1) & 8);      // +8 iff kk>=16
  return (((half << 4) | m) << 4) + e;     // lane*16 + e
}
// B matrix 32x16 bf16: same K order as A: K = (e<8 ? 8*half + e : 16 + 8*half + (e-8))
__device__ __forceinline__ int b_off(int kk, int n16) {
  int half = (kk >> 3) & 1;
  int e = (kk & 7) | ((kk >> 1) & 8);
  return (((half << 4) | n16) << 4) + e;
}

// ---------------------------------------------------------------------------
// Weight prep: fp32 -> bf16, pre-swizzled into B-fragment layout (512 bf16 /
// fragment = 32 lanes x 16 elems). Output layout (bf16 element offsets):
//   [0,32768)      edge W1 : t*16384 + (kb*4+nb)*512       (8x4 frags, 2 etypes)
//   [32768,36864)  edge W2 : 32768 + t*2048 + (kb*2+nb)*512
//   [36864,49152)  node W1 : 36864 + (kb*4+nb)*512          (6x4 frags)
//   [49152,57344)  node W2 : 49152 + (kb*8+nb)*512          (2x8 frags)
// ---------------------------------------------------------------------------
__global__ void gnn_prep_weights(const float* __restrict__ eW1, const float* __restrict__ eW2,
                                 const float* __restrict__ nW1, const float* __restrict__ nW2,
                                 int layer, bf16* __restrict__ out)
{
  int i = blockIdx.x * blockDim.x + threadIdx.x;
  if (i >= 57344) return;
  if (i < 32768) {                       // edge W1 [2][256][64]
    int t = i >> 14, r = i & 16383;
    int k = r >> 6, n = r & 63;
    float v = eW1[(((size_t)layer * 2 + t) * 256 + k) * 64 + n];
    int frag = ((k >> 5) << 2) + (n >> 4);
    out[t * 16384 + frag * 512 + b_off(k & 31, n & 15)] = (bf16)v;
  } else if (i < 36864) {                // edge W2 [2][64][32]
    int j = i - 32768;
    int t = j >> 11, r = j & 2047;
    int k = r >> 5, n = r & 31;
    float v = eW2[(((size_t)layer * 2 + t) * 64 + k) * 32 + n];
    int frag = ((k >> 5) << 1) + (n >> 4);
    out[32768 + t * 2048 + frag * 512 + b_off(k & 31, n & 15)] = (bf16)v;
  } else if (i < 49152) {                // node W1 [192][64]
    int j = i - 36864;
    int k = j >> 6, n = j & 63;
    float v = nW1[((size_t)layer * 192 + k) * 64 + n];
    int frag = ((k >> 5) << 2) + (n >> 4);
    out[36864 + frag * 512 + b_off(k & 31, n & 15)] = (bf16)v;
  } else {                               // node W2 [64][128]
    int j = i - 49152;
    int k = j >> 7, n = j & 127;
    float v = nW2[((size_t)layer * 64 + k) * 128 + n];
    int frag = ((k >> 5) << 3) + (n >> 4);
    out[49152 + frag * 512 + b_off(k & 31, n & 15)] = (bf16)v;
  }
}

// ---------------------------------------------------------------------------
// Edge kernel: 8 waves / block, 128 edges / block, 16 edges per wave (M rows).
// Both etypes computed per edge (reference semantics; mask applied at scatter).
// ---------------------------------------------------------------------------
__global__ __launch_bounds__(256) void gnn_edge_kernel(
    const float* __restrict__ h,
    const bf16* __restrict__ w1s, const bf16* __restrict__ w2s,
    const float* __restrict__ eb1, const float* __restrict__ eb2,
    const int* __restrict__ srcI, const int* __restrict__ dstI, const int* __restrict__ etI,
    float* __restrict__ msum, float* __restrict__ cnt)
{
  __shared__ __align__(32) bf16 aT[8][8][512];   // A fragments: 8 waves x 8 K-blocks
  __shared__ __align__(32) bf16 hT[8][2][512];   // hidden as A fragments for stage 2
  __shared__ int eDst[128];
  __shared__ int eEt[128];

  const int tid   = threadIdx.x;
  const int lane  = tid & 31;
  const int wv    = tid >> 5;
  const int laneN = lane & 15;
  const int laneH = lane >> 4;
  const int eBase = blockIdx.x * 128;

  // ---- gather: 2 threads per edge row (half 0 = src feats, half 1 = dst feats)
  {
    int row = tid >> 1;
    int half = tid & 1;
    int e = eBase + row;
    int dN = dstI[e];
    int node;
    if (half == 0) {
      int tp = etI[e];
      eDst[row] = dN; eEt[row] = tp;
      atomicAdd(&cnt[(size_t)dN * 2 + tp], 1.0f);     // segment count (mask sum)
      node = srcI[e];
    } else {
      node = dN;
    }
    const float* hp = h + (size_t)node * IND;
    int wrow = row >> 4, m = row & 15;
    #pragma unroll
    for (int i = 0; i < IND; i += 4) {
      float4 v = *(const float4*)(hp + i);
      int kd = half * IND + i;
      bf16* dp = &aT[wrow][kd >> 5][0];
      int kk = kd & 31;
      dp[a_off(kk + 0, m)] = (bf16)v.x;
      dp[a_off(kk + 1, m)] = (bf16)v.y;
      dp[a_off(kk + 2, m)] = (bf16)v.z;
      dp[a_off(kk + 3, m)] = (bf16)v.w;
    }
  }
  __syncthreads();

  for (int t = 0; t < 2; ++t) {
    // stage 1: [16,256] x [256,64]
    v8f acc[4] = {};
    #pragma unroll
    for (int kb = 0; kb < 8; ++kb) {
      v16bf a = *(const v16bf*)&aT[wv][kb][lane << 4];
      #pragma unroll
      for (int nb = 0; nb < 4; ++nb) {
        v16bf b = *(const v16bf*)&w1s[(((t * 8 + kb) << 2) + nb) * 512 + (lane << 4)];
        acc[nb] = __builtin_amdgcn_wmma_f32_16x16x32_bf16(
            false, a, false, b, (short)0, acc[nb], false, false);
      }
    }
    __syncthreads();                       // hT reused across t iterations
    // bias + ReLU, reshuffle hidden into A-fragment layout via LDS
    #pragma unroll
    for (int nb = 0; nb < 4; ++nb) {
      float bias = eb1[t * MH + nb * 16 + laneN];
      #pragma unroll
      for (int r = 0; r < 8; ++r) {
        float v = fmaxf(acc[nb][r] + bias, 0.0f);
        int m = r + (laneH << 3);
        int kd = nb * 16 + laneN;
        hT[wv][kd >> 5][a_off(kd & 31, m)] = (bf16)v;
      }
    }
    __syncthreads();
    // stage 2: [16,64] x [64,32]
    v8f acc2[2] = {};
    #pragma unroll
    for (int kb = 0; kb < 2; ++kb) {
      v16bf a = *(const v16bf*)&hT[wv][kb][lane << 4];
      #pragma unroll
      for (int nb = 0; nb < 2; ++nb) {
        v16bf b = *(const v16bf*)&w2s[(((t * 2 + kb) << 1) + nb) * 512 + (lane << 4)];
        acc2[nb] = __builtin_amdgcn_wmma_f32_16x16x32_bf16(
            false, a, false, b, (short)0, acc2[nb], false, false);
      }
    }
    // masked scatter (segment sum) — L2-resident float atomics
    #pragma unroll
    for (int nb = 0; nb < 2; ++nb) {
      float bias = eb2[t * EH + nb * 16 + laneN];
      #pragma unroll
      for (int r = 0; r < 8; ++r) {
        int m = r + (laneH << 3);
        int row = wv * 16 + m;
        if (eEt[row] == t) {
          atomicAdd(&msum[(size_t)eDst[row] * 64 + t * EH + nb * 16 + laneN],
                    acc2[nb][r] + bias);
        }
      }
    }
  }
}

// ---------------------------------------------------------------------------
// Node kernel: segment-mean on the fly + [N,192]x[192,64] -> ReLU -> x[64,128]
// ---------------------------------------------------------------------------
__global__ __launch_bounds__(256) void gnn_node_kernel(
    const float* __restrict__ h, const float* __restrict__ msum, const float* __restrict__ cnt,
    const bf16* __restrict__ w1s, const bf16* __restrict__ w2s,
    const float* __restrict__ b1, const float* __restrict__ b2,
    float* __restrict__ hout)
{
  __shared__ __align__(32) bf16 aT[8][6][512];
  __shared__ __align__(32) bf16 hT[8][2][512];

  const int tid   = threadIdx.x;
  const int lane  = tid & 31;
  const int wv    = tid >> 5;
  const int laneN = lane & 15;
  const int laneH = lane >> 4;
  const int nBase = blockIdx.x * 128;

  {
    int row = tid >> 1, half = tid & 1;
    int node = nBase + row; if (node >= NN) node = NN - 1;
    int wrow = row >> 4, m = row & 15;
    #pragma unroll
    for (int i = 0; i < 96; i += 4) {
      int c = half * 96 + i;           // 0..191 : [msg0|msg1|h]
      float4 v;
      if (c < 64) {
        v = *(const float4*)(msum + (size_t)node * 64 + c);
        float inv = 1.0f / fmaxf(cnt[(size_t)node * 2 + (c >> 5)], 1.0f);
        v.x *= inv; v.y *= inv; v.z *= inv; v.w *= inv;
      } else {
        v = *(const float4*)(h + (size_t)node * IND + (c - 64));
      }
      bf16* dp = &aT[wrow][c >> 5][0];
      int kk = c & 31;
      dp[a_off(kk + 0, m)] = (bf16)v.x;
      dp[a_off(kk + 1, m)] = (bf16)v.y;
      dp[a_off(kk + 2, m)] = (bf16)v.z;
      dp[a_off(kk + 3, m)] = (bf16)v.w;
    }
  }
  __syncthreads();

  v8f acc[4] = {};
  #pragma unroll
  for (int kb = 0; kb < 6; ++kb) {
    v16bf a = *(const v16bf*)&aT[wv][kb][lane << 4];
    #pragma unroll
    for (int nb = 0; nb < 4; ++nb) {
      v16bf b = *(const v16bf*)&w1s[((kb << 2) + nb) * 512 + (lane << 4)];
      acc[nb] = __builtin_amdgcn_wmma_f32_16x16x32_bf16(
          false, a, false, b, (short)0, acc[nb], false, false);
    }
  }
  #pragma unroll
  for (int nb = 0; nb < 4; ++nb) {
    float bias = b1[nb * 16 + laneN];
    #pragma unroll
    for (int r = 0; r < 8; ++r) {
      float v = fmaxf(acc[nb][r] + bias, 0.0f);
      int m = r + (laneH << 3);
      int kd = nb * 16 + laneN;
      hT[wv][kd >> 5][a_off(kd & 31, m)] = (bf16)v;
    }
  }
  __syncthreads();

  v8f acc2[8] = {};
  #pragma unroll
  for (int kb = 0; kb < 2; ++kb) {
    v16bf a = *(const v16bf*)&hT[wv][kb][lane << 4];
    #pragma unroll
    for (int nb = 0; nb < 8; ++nb) {
      v16bf b = *(const v16bf*)&w2s[((kb << 3) + nb) * 512 + (lane << 4)];
      acc2[nb] = __builtin_amdgcn_wmma_f32_16x16x32_bf16(
          false, a, false, b, (short)0, acc2[nb], false, false);
    }
  }
  #pragma unroll
  for (int nb = 0; nb < 8; ++nb) {
    float bias = b2[nb * 16 + laneN];
    #pragma unroll
    for (int r = 0; r < 8; ++r) {
      int m = r + (laneH << 3);
      int node = nBase + wv * 16 + m;
      if (node < NN)
        hout[(size_t)node * IND + nb * 16 + laneN] = acc2[nb][r] + bias;
    }
  }
}

// ---------------------------------------------------------------------------
extern "C" void kernel_launch(void* const* d_in, const int* in_sizes, int n_in,
                              void* d_out, int out_size, void* d_ws, size_t ws_size,
                              hipStream_t stream) {
  const float* nf   = (const float*)d_in[0];
  const float* eW1  = (const float*)d_in[1];
  const float* eb1  = (const float*)d_in[2];
  const float* eW2  = (const float*)d_in[3];
  const float* eb2  = (const float*)d_in[4];
  const float* nW1  = (const float*)d_in[5];
  const float* nb1  = (const float*)d_in[6];
  const float* nW2  = (const float*)d_in[7];
  const float* nb2  = (const float*)d_in[8];
  const int* src    = (const int*)d_in[9];
  const int* dst    = (const int*)d_in[10];
  const int* etype  = (const int*)d_in[11];

  float* h0   = (float*)d_ws;                       // [N,128]
  float* h1   = h0 + (size_t)NN * IND;              // [N,128]
  float* msum = h1 + (size_t)NN * IND;              // [N,64]
  float* cnt  = msum + (size_t)NN * 64;             // [N,2]
  bf16*  wswz = (bf16*)(cnt + (size_t)NN * 2);      // 57344 bf16 swizzled weights

  const float* hin = nf;
  float* houts[3] = { h0, h1, (float*)d_out };

  for (int l = 0; l < 3; ++l) {
    gnn_prep_weights<<<224, 256, 0, stream>>>(eW1, eW2, nW1, nW2, l, wswz);
    hipMemsetAsync(msum, 0, (size_t)NN * 64 * sizeof(float), stream);
    hipMemsetAsync(cnt,  0, (size_t)NN * 2 * sizeof(float), stream);
    gnn_edge_kernel<<<NE / 128, 256, 0, stream>>>(
        hin, wswz, wswz + 32768, eb1 + l * 2 * MH, eb2 + l * 2 * EH,
        src, dst, etype, msum, cnt);
    gnn_node_kernel<<<(NN + 127) / 128, 256, 0, stream>>>(
        hin, msum, cnt, wswz + 36864, wswz + 49152,
        nb1 + l * MH, nb2 + l * IND, houts[l]);
    hin = houts[l];
  }
}
